// PyGGCN_1382979469690
// MI455X (gfx1250) — matrix operand order repaired and checked by reading, not compiled, not run on hardware
//
#include <hip/hip_runtime.h>
#include <hip/hip_bf16.h>

// ---------------------------------------------------------------------------
// Types for CDNA5 WMMA (wave32)
// ---------------------------------------------------------------------------
typedef __attribute__((ext_vector_type(16))) __bf16 v16bf;
typedef __attribute__((ext_vector_type(8)))  __bf16 v8bf;
typedef __attribute__((ext_vector_type(8)))  float  v8f;

__device__ __forceinline__ __bf16 to_bf16(float f) {
    unsigned u = __float_as_uint(f);
    u += 0x7FFFu + ((u >> 16) & 1u);          // round-to-nearest-even
    unsigned short h = (unsigned short)(u >> 16);
    __bf16 r;
    __builtin_memcpy(&r, &h, 2);
    return r;
}

// ---------------------------------------------------------------------------
// Zero-fill (float4 granularity; all our buffers are multiples of 4 floats)
// ---------------------------------------------------------------------------
__global__ __launch_bounds__(256) void fill_zero4(float4* __restrict__ p, long long n4) {
    long long gid = (long long)blockIdx.x * blockDim.x + threadIdx.x;
    if (gid < n4) p[gid] = make_float4(0.f, 0.f, 0.f, 0.f);
}

// ---------------------------------------------------------------------------
// WMMA GEMM: C[nrows x DOUT] = act(A[nrows x DIN]) * W[DIN x DOUT]
// bf16 multiply, f32 accumulate. Block = 256 threads = 8 waves, 128 rows/block.
// Wave w computes rows [16w,16w+16), all DOUT/16 column tiles.
// ---------------------------------------------------------------------------
template <int DIN, int DOUT, bool RELU_IN>
__global__ __launch_bounds__(256) void gemm_wmma(const float* __restrict__ A,
                                                 const float* __restrict__ W,
                                                 float* __restrict__ C,
                                                 int nrows) {
    constexpr int AST = DIN + 16;   // padded LDS stride (bf16 elems) -> spreads banks
    constexpr int WST = DIN + 16;
    __shared__ __align__(16) __bf16 lds_a[128 * AST];      // A tile, row-major
    __shared__ __align__(16) __bf16 lds_w[DOUT * WST];     // W^T : [n][k]

    const int tid  = threadIdx.x;
    const int row0 = blockIdx.x * 128;

    // ---- stage A tile (float4 coalesced, fused ReLU, bf16 convert) ----
    #pragma unroll 2
    for (int idx = tid; idx < 128 * (DIN / 4); idx += 256) {
        const int r  = idx / (DIN / 4);
        const int c4 = (idx % (DIN / 4)) * 4;
        float4 v = make_float4(0.f, 0.f, 0.f, 0.f);
        const int gr = row0 + r;
        if (gr < nrows) v = *reinterpret_cast<const float4*>(A + (size_t)gr * DIN + c4);
        if (RELU_IN) {
            v.x = fmaxf(v.x, 0.f); v.y = fmaxf(v.y, 0.f);
            v.z = fmaxf(v.z, 0.f); v.w = fmaxf(v.w, 0.f);
        }
        __bf16* p = &lds_a[r * AST + c4];
        p[0] = to_bf16(v.x); p[1] = to_bf16(v.y);
        p[2] = to_bf16(v.z); p[3] = to_bf16(v.w);
    }
    // ---- stage W transposed ----
    for (int idx = tid; idx < DIN * DOUT; idx += 256) {
        const int k = idx / DOUT;
        const int n = idx % DOUT;
        lds_w[n * WST + k] = to_bf16(W[idx]);
    }
    __syncthreads();

    const int lane = tid & 31;
    const int wave = tid >> 5;
    const int rr   = lane & 15;   // row-in-tile (and column-in-tile for B/C)
    const int half = lane >> 4;   // lane half selects K sub-range

    constexpr int NT = DOUT / 16;
    v8f acc[NT];
    #pragma unroll
    for (int t = 0; t < NT; ++t) acc[t] = (v8f){0.f,0.f,0.f,0.f,0.f,0.f,0.f,0.f};

    const int arow = wave * 16 + rr;

    #pragma unroll
    for (int kk = 0; kk < DIN; kk += 32) {
        // A fragment (ISA 16-bit A 16x32 layout):
        // lanes 0-15:  K = kk+0..7  and kk+16..23 ; lanes 16-31: +8
        v8bf alo = *(const v8bf*)(&lds_a[arow * AST + kk + half * 8]);
        v8bf ahi = *(const v8bf*)(&lds_a[arow * AST + kk + 16 + half * 8]);
        v16bf a;
        #pragma unroll
        for (int i = 0; i < 8; ++i) { a[i] = alo[i]; a[8 + i] = ahi[i]; }

        #pragma unroll
        for (int t = 0; t < NT; ++t) {
            // B fragment: column n = rr, lanes 0-15: K=kk+0..7 and kk+16..23, lanes 16-31: +8
            const int n = t * 16 + rr;
            v8bf b0 = *(const v8bf*)(&lds_w[n * WST + kk + half * 8]);
            v8bf b1 = *(const v8bf*)(&lds_w[n * WST + kk + 16 + half * 8]);
            v16bf b;
            #pragma unroll
            for (int i = 0; i < 8; ++i) { b[i] = b0[i]; b[8 + i] = b1[i]; }

            acc[t] = __builtin_amdgcn_wmma_f32_16x16x32_bf16(
                         false, a, false, b, (short)0, acc[t], false, false);
        }
    }

    // ---- store C: element i of v8f -> row = half*8+i, col = rr ----
    const int grow0 = row0 + wave * 16 + half * 8;
    #pragma unroll
    for (int t = 0; t < NT; ++t) {
        #pragma unroll
        for (int i = 0; i < 8; ++i) {
            const int gr = grow0 + i;
            if (gr < nrows) C[(size_t)gr * DOUT + t * 16 + rr] = acc[t][i];
        }
    }
}

// ---------------------------------------------------------------------------
// Edge scatter-add: out[dst[e], :] += msg[src[e], :]
// DIM=128: one wave per edge (lane handles float4). DIM=64: half wave per edge.
// ---------------------------------------------------------------------------
template <int DIM>
__global__ __launch_bounds__(256) void scatter_add(const float* __restrict__ msg,
                                                   const int* __restrict__ src,
                                                   const int* __restrict__ dst,
                                                   float* __restrict__ out,
                                                   int nedges) {
    constexpr int CHUNK = DIM / 4;
    const long long gid = (long long)blockIdx.x * blockDim.x + threadIdx.x;
    const long long e   = gid / CHUNK;
    if (e >= nedges) return;
    const int c = (int)(gid % CHUNK) * 4;
    const int s = src[e];
    const int d = dst[e];
    const float4 v = *reinterpret_cast<const float4*>(msg + (size_t)s * DIM + c);
    float* o = out + (size_t)d * DIM + c;
    unsafeAtomicAdd(o + 0, v.x);
    unsafeAtomicAdd(o + 1, v.y);
    unsafeAtomicAdd(o + 2, v.z);
    unsafeAtomicAdd(o + 3, v.w);
}

// ---------------------------------------------------------------------------
// In-place log_softmax over rows of 64 (one wave32 per row, 2 elems/lane)
// ---------------------------------------------------------------------------
__global__ __launch_bounds__(256) void log_softmax64(float* __restrict__ out, int nrows) {
    const long long gwave = ((long long)blockIdx.x * blockDim.x + threadIdx.x) >> 5;
    const int lane = threadIdx.x & 31;
    if (gwave >= nrows) return;                     // wave-uniform exit
    float* row = out + gwave * 64;
    float a = row[lane];
    float b = row[32 + lane];
    float m = fmaxf(a, b);
    #pragma unroll
    for (int off = 16; off > 0; off >>= 1) m = fmaxf(m, __shfl_xor(m, off, 32));
    float s = __expf(a - m) + __expf(b - m);
    #pragma unroll
    for (int off = 16; off > 0; off >>= 1) s += __shfl_xor(s, off, 32);
    const float ls = m + __logf(s);
    row[lane]      = a - ls;
    row[32 + lane] = b - ls;
}

// ---------------------------------------------------------------------------
// Launch: x, edge_index, W1, W2, W3  ->  log_softmax(GCN3(x))
// ---------------------------------------------------------------------------
extern "C" void kernel_launch(void* const* d_in, const int* in_sizes, int n_in,
                              void* d_out, int out_size, void* d_ws, size_t ws_size,
                              hipStream_t stream) {
    const float* x  = (const float*)d_in[0];
    const int*   ei = (const int*)d_in[1];
    const float* W1 = (const float*)d_in[2];
    const float* W2 = (const float*)d_in[3];
    const float* W3 = (const float*)d_in[4];

    const int N = in_sizes[0] / 128;   // 100000
    const int E = in_sizes[1] / 2;     // 1600000
    const int* src = ei;
    const int* dst = ei + E;

    float* tmp  = (float*)d_ws;                 // N x 128 message buffer
    float* hbuf = tmp + (size_t)N * 128;        // N x 128 aggregation buffer
    float* outf = (float*)d_out;                // N x 64

    const int gemmGrid = (N + 127) / 128;
    const long long z128 = (long long)N * 128 / 4;
    const long long z64  = (long long)N * 64 / 4;
    const int zg128 = (int)((z128 + 255) / 256);
    const int zg64  = (int)((z64 + 255) / 256);
    const int sg128 = (int)(((long long)E * 32 + 255) / 256);
    const int sg64  = (int)(((long long)E * 16 + 255) / 256);
    const int lsg   = (int)(((long long)N * 32 + 255) / 256);

    // Layer 1: tmp = x @ W1 ; h = scatter(tmp)
    fill_zero4<<<zg128, 256, 0, stream>>>((float4*)hbuf, z128);
    gemm_wmma<128, 128, false><<<gemmGrid, 256, 0, stream>>>(x, W1, tmp, N);
    scatter_add<128><<<sg128, 256, 0, stream>>>(tmp, src, dst, hbuf, E);

    // Layer 2: tmp = relu(h) @ W2 ; h = scatter(tmp)   (hbuf reused after gemm reads it)
    gemm_wmma<128, 128, true><<<gemmGrid, 256, 0, stream>>>(hbuf, W2, tmp, N);
    fill_zero4<<<zg128, 256, 0, stream>>>((float4*)hbuf, z128);
    scatter_add<128><<<sg128, 256, 0, stream>>>(tmp, src, dst, hbuf, E);

    // Layer 3: tmp = relu(h) @ W3 ; out = scatter(tmp) ; log_softmax in place
    gemm_wmma<128, 64, true><<<gemmGrid, 256, 0, stream>>>(hbuf, W3, tmp, N);
    fill_zero4<<<zg64, 256, 0, stream>>>((float4*)outf, z64);
    scatter_add<64><<<sg64, 256, 0, stream>>>(tmp, src, dst, outf, E);
    log_softmax64<<<lsg, 256, 0, stream>>>(outf, N);
}
